// TransformerLayer_33681133535784
// MI455X (gfx1250) — compile-verified, matrix operand order checked by reading, not run on hardware
//
#include <hip/hip_runtime.h>
#include <hip/hip_bf16.h>

// ---------------------------------------------------------------------------
// Types matching CDNA5 WMMA builtin signatures
// ---------------------------------------------------------------------------
typedef __attribute__((ext_vector_type(16))) __bf16 v16bf;
typedef __attribute__((ext_vector_type(8)))  float  v8f;
typedef __attribute__((ext_vector_type(4)))  int    v4i;

__device__ __forceinline__ v8f wmma_bf16(v16bf a, v16bf b, v8f c) {
  return __builtin_amdgcn_wmma_f32_16x16x32_bf16(
      /*neg_a=*/false, a, /*neg_b=*/false, b,
      /*c_mod=*/(short)0, c, /*reuse_a=*/false, /*reuse_b=*/false);
}

// ---------------------------------------------------------------------------
// CDNA5 async global->LDS copy (ASYNCcnt path), with sync fallback.
// Builtin signature (verified by toolchain diagnostic):
//   (as1 v4i* gsrc, as3 v4i* ldst, int offset, int cpol)
// ---------------------------------------------------------------------------
#if defined(__AMDGCN__) && __has_builtin(__builtin_amdgcn_global_load_async_to_lds_b128)
#define HAVE_ASYNC 1
typedef __attribute__((address_space(1))) v4i* gas_v4i_ptr;
typedef __attribute__((address_space(3))) v4i* las_v4i_ptr;
#else
#define HAVE_ASYNC 0
#endif

__device__ __forceinline__ void copy16(const __bf16* g, __bf16* l) {
#if HAVE_ASYNC
  __builtin_amdgcn_global_load_async_to_lds_b128(
      (gas_v4i_ptr)(unsigned long long)g,
      (las_v4i_ptr)(unsigned int)(unsigned long long)l,
      0, 0);
#else
  *reinterpret_cast<uint4*>(l) = *reinterpret_cast<const uint4*>(g);
#endif
}

template <int N>
__device__ __forceinline__ void wait_async() {
#if HAVE_ASYNC
#if __has_builtin(__builtin_amdgcn_s_wait_asynccnt)
  __builtin_amdgcn_s_wait_asynccnt(N);
#else
  asm volatile("s_wait_asynccnt %0" ::"i"(N) : "memory");
#endif
#endif
}

// ---------------------------------------------------------------------------
// 16x32 bf16 fragment load from LDS (ISA 16-bit A layout; B tiles stored [N][K]
// use the same layout). Compiler merges these into ds_load_b128.
// ---------------------------------------------------------------------------
__device__ __forceinline__ v16bf ld_frag(const __bf16* base, int stride) {
  const int lane = threadIdx.x & 31;
  const int r    = lane & 15;
  const int kb   = (lane >> 4) << 3;   // 0 or 8
  const __bf16* p = base + r * stride + kb;
  v16bf f;
#pragma unroll
  for (int i = 0; i < 4; ++i) {        // VGPRs 0..3: K = kb + {0..7}
    f[2 * i]     = p[2 * i];
    f[2 * i + 1] = p[2 * i + 1];
  }
#pragma unroll
  for (int i = 0; i < 4; ++i) {        // VGPRs 4..7: K = kb + 16 + {0..7}
    f[8 + 2 * i] = p[16 + 2 * i];
    f[9 + 2 * i] = p[17 + 2 * i];
  }
  return f;
}

// ---------------------------------------------------------------------------
// f32 -> bf16 cast (activations)
// ---------------------------------------------------------------------------
__global__ __launch_bounds__(256) void cast_f32_bf16(const float* __restrict__ s,
                                                     __bf16* __restrict__ d, int n) {
  int i = blockIdx.x * 256 + threadIdx.x;
  if (i < n) d[i] = (__bf16)s[i];
}

// ---------------------------------------------------------------------------
// Weight cast + transpose: w[K][N] f32 -> wt[N][K] bf16 (tiled via LDS)
// ---------------------------------------------------------------------------
__global__ __launch_bounds__(256) void cast_transpose(const float* __restrict__ w,
                                                      __bf16* __restrict__ wt,
                                                      int K, int N) {
  __shared__ float tile[32][33];
  const int k0 = blockIdx.y * 32, n0 = blockIdx.x * 32;
  const int tx = threadIdx.x & 31, ty = threadIdx.x >> 5;
#pragma unroll
  for (int i = 0; i < 32; i += 8)
    tile[ty + i][tx] = w[(size_t)(k0 + ty + i) * N + n0 + tx];
  __syncthreads();
#pragma unroll
  for (int i = 0; i < 32; i += 8)
    wt[(size_t)(n0 + ty + i) * K + k0 + tx] = (__bf16)tile[tx][ty + i];
}

// ---------------------------------------------------------------------------
// V pre-transpose: qkv[(b*2048+s)][2048 + h*64 + d] -> vt[(bh*64+d)*2048 + s]
// Makes V^T tiles contiguous so attention can stage them asynchronously.
// Grid (2048/32, 64/32, 32); block 256 = 32x8.
// ---------------------------------------------------------------------------
__global__ __launch_bounds__(256) void transpose_v(const __bf16* __restrict__ qkv,
                                                   __bf16* __restrict__ vt) {
  __shared__ __bf16 tile[32][33];
  const int bh = blockIdx.z;
  const int b = bh >> 4, h = bh & 15;
  const int s0 = blockIdx.x * 32;
  const int d0 = blockIdx.y * 32;
  const int tx = threadIdx.x & 31, ty = threadIdx.x >> 5;
#pragma unroll
  for (int i = 0; i < 32; i += 8)
    tile[ty + i][tx] =
        qkv[(size_t)(b * 2048 + s0 + ty + i) * 3072 + 2048 + h * 64 + d0 + tx];
  __syncthreads();
#pragma unroll
  for (int i = 0; i < 32; i += 8)
    vt[((size_t)bh * 64 + d0 + ty + i) * 2048 + s0 + tx] = tile[tx][ty + i];
}

// ---------------------------------------------------------------------------
// bf16 GEMM: C[M,N] = A[M,K] * Bt[N,K]^T (+bias)(gelu?) -> f32 / bf16
// 256 threads (8 wave32), tile 128x128, BK=64, double-buffered LDS with
// async global->LDS staging. Wave w: rows (w&3)*32, cols (w>>2)*64.
// ---------------------------------------------------------------------------
#define BM 128
#define BN 128
#define BK 64
#define LDA (BK + 8)

__global__ __launch_bounds__(256) void gemm_bf16_kernel(
    const __bf16* __restrict__ A, const __bf16* __restrict__ Bt,
    const float* __restrict__ bias, float* __restrict__ Cf,
    __bf16* __restrict__ Cb, int M, int N, int K, int gelu) {
  __shared__ __bf16 sA[2][BM][LDA];
  __shared__ __bf16 sB[2][BN][LDA];

  const int tid  = threadIdx.x;
  const int lane = tid & 31;
  const int wave = tid >> 5;
  const int wm   = wave & 3;
  const int wn   = wave >> 2;
  const int m0   = blockIdx.y * BM;
  const int n0   = blockIdx.x * BN;

  v8f acc[2][4] = {};
  const int nt = K / BK;

  auto issue = [&](int it) {
    const int s  = it & 1;
    const int k0 = it * BK;
    const int base = tid * 4;
#pragma unroll
    for (int c = 0; c < 4; ++c) {
      const int id = base + c;            // 0..1023
      const int row = id >> 3, ch = id & 7;
      copy16(A + (size_t)(m0 + row) * K + k0 + ch * 8, &sA[s][row][ch * 8]);
    }
#pragma unroll
    for (int c = 0; c < 4; ++c) {
      const int id = base + c;
      const int row = id >> 3, ch = id & 7;
      copy16(Bt + (size_t)(n0 + row) * K + k0 + ch * 8, &sB[s][row][ch * 8]);
    }
  };

  issue(0);
  for (int it = 0; it < nt; ++it) {
    if (it + 1 < nt) {
      issue(it + 1);       // next tile in flight behind this tile's WMMAs
      wait_async<8>();     // current tile complete; 8 ops (next tile) pending
    } else {
      wait_async<0>();
    }
    __syncthreads();

    const int s = it & 1;
#pragma unroll
    for (int kk = 0; kk < BK; kk += 32) {
      v16bf aF[2], bF[4];
#pragma unroll
      for (int mi = 0; mi < 2; ++mi)
        aF[mi] = ld_frag(&sA[s][wm * 32 + mi * 16][kk], LDA);
#pragma unroll
      for (int ni = 0; ni < 4; ++ni)
        bF[ni] = ld_frag(&sB[s][wn * 64 + ni * 16][kk], LDA);
#pragma unroll
      for (int mi = 0; mi < 2; ++mi)
#pragma unroll
        for (int ni = 0; ni < 4; ++ni)
          acc[mi][ni] = wmma_bf16(aF[mi], bF[ni], acc[mi][ni]);
    }
    __syncthreads();
  }

#pragma unroll
  for (int mi = 0; mi < 2; ++mi) {
#pragma unroll
    for (int ni = 0; ni < 4; ++ni) {
#pragma unroll
      for (int e = 0; e < 8; ++e) {
        const int row = m0 + wm * 32 + mi * 16 + e + 8 * (lane >> 4);
        const int col = n0 + wn * 64 + ni * 16 + (lane & 15);
        float v = acc[mi][ni][e];
        if (bias) v += bias[col];
        if (gelu) v = 0.5f * v * (1.0f + erff(v * 0.70710678118654752f));
        if (Cf) Cf[(size_t)row * N + col] = v;
        if (Cb) Cb[(size_t)row * N + col] = (__bf16)v;
      }
    }
  }
}

// ---------------------------------------------------------------------------
// Flash attention: qkv bf16 [B*S][3072] + vt bf16 [32][64][2048] -> out bf16
// Grid: (S/128, B*H). 256 threads = 8 waves; wave owns 16 query rows.
// Double-buffered async K/V^T staging; QK^T and P*V via WMMA; online softmax.
// ---------------------------------------------------------------------------
#define AQ  128
#define AKC 64
#define QLD 72           // 64 + 8 pad (bf16)
#define SLD 65           // 64 + 1 pad (f32)
#define KVSTRIDE (64 * QLD)   // one K or V^T stage, bf16 elems

#define OFF_Q   0                     // [128][72]   bf16 = 18432
#define OFF_K   18432                 // [2][64][72] bf16 = 18432
#define OFF_VT  36864                 // [2][64][72] bf16 = 18432
#define OFF_S   55296                 // [8][16][65] f32  = 33280
#define OFF_P   88576                 // [8][16][72] bf16 = 18432
#define OFF_ST  107008                // [8][16][3]  f32  =  1536
#define ATTN_LDS_BYTES 108544

__global__ __launch_bounds__(256) void attn_kernel(const __bf16* __restrict__ qkv,
                                                   const __bf16* __restrict__ vtg,
                                                   __bf16* __restrict__ out) {
  extern __shared__ char smem[];
  __bf16* sQ  = reinterpret_cast<__bf16*>(smem + OFF_Q);
  __bf16* sK  = reinterpret_cast<__bf16*>(smem + OFF_K);
  __bf16* sVT = reinterpret_cast<__bf16*>(smem + OFF_VT);
  float*  sS  = reinterpret_cast<float*>(smem + OFF_S);
  __bf16* sP  = reinterpret_cast<__bf16*>(smem + OFF_P);
  float*  sSt = reinterpret_cast<float*>(smem + OFF_ST);

  const int tid  = threadIdx.x;
  const int lane = tid & 31;
  const int wave = tid >> 5;
  const int bh   = blockIdx.y;
  const int b    = bh >> 4;
  const int h    = bh & 15;
  const int q0   = blockIdx.x * AQ;
  const size_t rs = 3 * 1024;

  const __bf16* qb = qkv + (size_t)(b * 2048) * rs + h * 64;
  const __bf16* kb = qb + 1024;
  const __bf16* vt = vtg + (size_t)bh * 64 * 2048;   // [d][s]

  // Q tile, pre-scaled by 1/sqrt(64)
  for (int idx = tid; idx < AQ * 64; idx += 256) {
    const int r = idx >> 6, d = idx & 63;
    sQ[r * QLD + d] = (__bf16)((float)qb[(size_t)(q0 + r) * rs + d] * 0.125f);
  }
  if (lane < 16) {
    sSt[(wave * 16 + lane) * 3 + 0] = -1e30f;
    sSt[(wave * 16 + lane) * 3 + 1] = 0.0f;
  }
  v8f oacc[4] = {};

  // async stage of one 64-key chunk: K rows + V^T rows, 4 b128 per thread
  auto issueKV = [&](int it) {
    const int s  = it & 1;
    const int kc = it * AKC;
    const int base = tid * 2;
    __bf16* dK = sK + s * KVSTRIDE;
    __bf16* dV = sVT + s * KVSTRIDE;
#pragma unroll
    for (int c = 0; c < 2; ++c) {
      const int id = base + c;            // 0..511
      const int r = id >> 3, ch = id & 7;
      copy16(kb + (size_t)(kc + r) * rs + ch * 8, dK + r * QLD + ch * 8);
    }
#pragma unroll
    for (int c = 0; c < 2; ++c) {
      const int id = base + c;
      const int r = id >> 3, ch = id & 7;  // r = head-dim row of V^T
      copy16(vt + (size_t)r * 2048 + kc + ch * 8, dV + r * QLD + ch * 8);
    }
  };

  issueKV(0);
  __syncthreads();   // covers sQ/sSt init too

  for (int it = 0; it < 2048 / AKC; ++it) {
    if (it + 1 < 2048 / AKC) {
      issueKV(it + 1);   // next chunk in flight behind this chunk's WMMAs
      wait_async<4>();
    } else {
      wait_async<0>();
    }
    __syncthreads();

    const int s = it & 1;
    const __bf16* kBase = sK + s * KVSTRIDE;
    const __bf16* vBase = sVT + s * KVSTRIDE;

    // S = Q K^T
    v8f sacc[4] = {};
    const v16bf a0 = ld_frag(&sQ[(wave * 16) * QLD + 0], QLD);
    const v16bf a1 = ld_frag(&sQ[(wave * 16) * QLD + 32], QLD);
#pragma unroll
    for (int ni = 0; ni < 4; ++ni) {
      const v16bf b0 = ld_frag(&kBase[(ni * 16) * QLD + 0], QLD);
      const v16bf b1 = ld_frag(&kBase[(ni * 16) * QLD + 32], QLD);
      sacc[ni] = wmma_bf16(a0, b0, sacc[ni]);
      sacc[ni] = wmma_bf16(a1, b1, sacc[ni]);
    }
#pragma unroll
    for (int ni = 0; ni < 4; ++ni)
#pragma unroll
      for (int e = 0; e < 8; ++e)
        sS[(wave * 16 + e + 8 * (lane >> 4)) * SLD + ni * 16 + (lane & 15)] =
            sacc[ni][e];

    // online softmax: lanes 0..15 own one query row each (same-wave ordering)
    if (lane < 16) {
      float* st = &sSt[(wave * 16 + lane) * 3];
      const float* sr = &sS[(wave * 16 + lane) * SLD];
      __bf16* pr = &sP[(wave * 16 + lane) * QLD];
      float m_old = st[0], l = st[1];
      float mx = m_old;
      for (int j = 0; j < AKC; ++j) mx = fmaxf(mx, sr[j]);
      const float scale = __expf(m_old - mx);
      l *= scale;
      for (int j = 0; j < AKC; ++j) {
        const float p = __expf(sr[j] - mx);
        pr[j] = (__bf16)p;
        l += p;
      }
      st[0] = mx; st[1] = l; st[2] = scale;
    }

#pragma unroll
    for (int ni = 0; ni < 4; ++ni)
#pragma unroll
      for (int e = 0; e < 8; ++e) {
        const int r = e + 8 * (lane >> 4);
        oacc[ni][e] *= sSt[(wave * 16 + r) * 3 + 2];
      }

    // O += P V
    const v16bf p0 = ld_frag(&sP[(wave * 16) * QLD + 0], QLD);
    const v16bf p1 = ld_frag(&sP[(wave * 16) * QLD + 32], QLD);
#pragma unroll
    for (int ni = 0; ni < 4; ++ni) {
      const v16bf v0 = ld_frag(&vBase[(ni * 16) * QLD + 0], QLD);
      const v16bf v1 = ld_frag(&vBase[(ni * 16) * QLD + 32], QLD);
      oacc[ni] = wmma_bf16(p0, v0, oacc[ni]);
      oacc[ni] = wmma_bf16(p1, v1, oacc[ni]);
    }
    __syncthreads();
  }

#pragma unroll
  for (int ni = 0; ni < 4; ++ni)
#pragma unroll
    for (int e = 0; e < 8; ++e) {
      const int r = e + 8 * (lane >> 4);
      const float l = sSt[(wave * 16 + r) * 3 + 1];
      const int row = q0 + wave * 16 + r;
      const int d   = ni * 16 + (lane & 15);
      out[(size_t)(b * 2048 + row) * 1024 + h * 64 + d] = (__bf16)(oacc[ni][e] / l);
    }
}

// ---------------------------------------------------------------------------
// Residual + LayerNorm over rows of 1024; block per row; float4 traffic.
// ---------------------------------------------------------------------------
__global__ __launch_bounds__(256) void ln_residual_kernel(
    const float* __restrict__ a, const float* __restrict__ r,
    const float* __restrict__ gam, const float* __restrict__ bet,
    float* __restrict__ of, __bf16* __restrict__ ob) {
  const int row = blockIdx.x;
  const int tid = threadIdx.x;
  __shared__ float red[256];
  const float4* pa = (const float4*)(a + (size_t)row * 1024);
  const float4* pr = (const float4*)(r + (size_t)row * 1024);

  const float4 a4 = pa[tid];
  const float4 r4 = pr[tid];
  float v[4] = {a4.x + r4.x, a4.y + r4.y, a4.z + r4.z, a4.w + r4.w};
  float s = v[0] + v[1] + v[2] + v[3];

  red[tid] = s; __syncthreads();
  for (int off = 128; off > 0; off >>= 1) {
    if (tid < off) red[tid] += red[tid + off];
    __syncthreads();
  }
  const float mu = red[0] * (1.0f / 1024.0f);
  __syncthreads();
  s = 0.f;
#pragma unroll
  for (int i = 0; i < 4; ++i) { const float d = v[i] - mu; s += d * d; }
  red[tid] = s; __syncthreads();
  for (int off = 128; off > 0; off >>= 1) {
    if (tid < off) red[tid] += red[tid + off];
    __syncthreads();
  }
  const float rstd = rsqrtf(red[0] * (1.0f / 1024.0f) + 1e-5f);

  const float4 g4 = ((const float4*)gam)[tid];
  const float4 b4 = ((const float4*)bet)[tid];
  float y[4];
  y[0] = (v[0] - mu) * rstd * g4.x + b4.x;
  y[1] = (v[1] - mu) * rstd * g4.y + b4.y;
  y[2] = (v[2] - mu) * rstd * g4.z + b4.z;
  y[3] = (v[3] - mu) * rstd * g4.w + b4.w;
  ((float4*)(of + (size_t)row * 1024))[tid] = make_float4(y[0], y[1], y[2], y[3]);
  if (ob) {
    __bf16* po = ob + (size_t)row * 1024 + tid * 4;
#pragma unroll
    for (int i = 0; i < 4; ++i) po[i] = (__bf16)y[i];
  }
}

// ---------------------------------------------------------------------------
// Host orchestration
// ---------------------------------------------------------------------------
static inline size_t align256(size_t x) { return (x + 255) & ~(size_t)255; }

extern "C" void kernel_launch(void* const* d_in, const int* in_sizes, int n_in,
                              void* d_out, int out_size, void* d_ws, size_t ws_size,
                              hipStream_t stream) {
  (void)in_sizes; (void)n_in; (void)out_size; (void)ws_size;
  const float* x     = (const float*)d_in[0];
  const float* w_qkv = (const float*)d_in[1];
  const float* b_qkv = (const float*)d_in[2];
  const float* w_out = (const float*)d_in[3];
  const float* b_out = (const float*)d_in[4];
  const float* w_ff1 = (const float*)d_in[5];
  const float* b_ff1 = (const float*)d_in[6];
  const float* w_ff2 = (const float*)d_in[7];
  const float* b_ff2 = (const float*)d_in[8];
  const float* ln1_g = (const float*)d_in[9];
  const float* ln1_b = (const float*)d_in[10];
  const float* ln2_g = (const float*)d_in[11];
  const float* ln2_b = (const float*)d_in[12];

  const int T = 2 * 2048;
  const int D = 1024, DQKV = 3072, DFF = 4096;

  char* ws = (char*)d_ws;
  size_t off = 0;
  auto carve = [&](size_t bytes) { char* p = ws + off; off = align256(off + bytes); return p; };

  __bf16* xb    = (__bf16*)carve((size_t)T * D * 2);
  __bf16* wqkvt = (__bf16*)carve((size_t)D * DQKV * 2);   // [3072][1024]
  __bf16* woutt = (__bf16*)carve((size_t)D * D * 2);      // [1024][1024]
  __bf16* wff1t = (__bf16*)carve((size_t)D * DFF * 2);    // [4096][1024]
  __bf16* wff2t = (__bf16*)carve((size_t)DFF * D * 2);    // [1024][4096]
  __bf16* qkvb  = (__bf16*)carve((size_t)T * DQKV * 2);
  __bf16* vtb   = (__bf16*)carve((size_t)32 * 64 * 2048 * 2);  // V^T per head
  __bf16* ob    = (__bf16*)carve((size_t)T * D * 2);
  float*  attnf = (float*)carve((size_t)T * D * 4);
  float*  x1f   = (float*)carve((size_t)T * D * 4);
  __bf16* x1b   = (__bf16*)carve((size_t)T * D * 2);
  __bf16* hb    = (__bf16*)carve((size_t)T * DFF * 2);
  float*  fff   = (float*)carve((size_t)T * D * 4);

  cast_f32_bf16<<<(T * D + 255) / 256, 256, 0, stream>>>(x, xb, T * D);
  cast_transpose<<<dim3(DQKV / 32, D / 32), 256, 0, stream>>>(w_qkv, wqkvt, D, DQKV);
  cast_transpose<<<dim3(D / 32, D / 32), 256, 0, stream>>>(w_out, woutt, D, D);
  cast_transpose<<<dim3(DFF / 32, D / 32), 256, 0, stream>>>(w_ff1, wff1t, D, DFF);
  cast_transpose<<<dim3(D / 32, DFF / 32), 256, 0, stream>>>(w_ff2, wff2t, DFF, D);

  // 1) QKV projection (bias fused) -> bf16
  gemm_bf16_kernel<<<dim3(DQKV / BN, T / BM), 256, 0, stream>>>(
      xb, wqkvt, b_qkv, nullptr, qkvb, T, DQKV, D, 0);

  // 1b) pre-transpose V per head for async attention staging
  transpose_v<<<dim3(2048 / 32, 64 / 32, 32), 256, 0, stream>>>(qkvb, vtb);

  // 2) attention -> ob bf16
  attn_kernel<<<dim3(2048 / AQ, 2 * 16), 256, ATTN_LDS_BYTES, stream>>>(qkvb, vtb, ob);

  // 3) out projection -> f32
  gemm_bf16_kernel<<<dim3(D / BN, T / BM), 256, 0, stream>>>(
      ob, woutt, b_out, attnf, nullptr, T, D, D, 0);

  // 4) x1 = LN(x + attn_out)
  ln_residual_kernel<<<T, 256, 0, stream>>>(attnf, x, ln1_g, ln1_b, x1f, x1b);

  // 5) FF1 + exact GELU -> bf16
  gemm_bf16_kernel<<<dim3(DFF / BN, T / BM), 256, 0, stream>>>(
      x1b, wff1t, b_ff1, nullptr, hb, T, DFF, D, 1);

  // 6) FF2 -> f32
  gemm_bf16_kernel<<<dim3(D / BN, T / BM), 256, 0, stream>>>(
      hb, wff2t, b_ff2, fff, nullptr, T, D, DFF, 0);

  // 7) out = LN(x1 + ff)
  ln_residual_kernel<<<T, 256, 0, stream>>>(fff, x1f, ln2_g, ln2_b, (float*)d_out, nullptr);
}
